// LSTM_80298708566551
// MI455X (gfx1250) — matrix operand order repaired and checked by reading, not compiled, not run on hardware
//
#include <hip/hip_runtime.h>
#include <hip/hip_bf16.h>

typedef __attribute__((ext_vector_type(16))) __bf16        v16bf;
typedef __attribute__((ext_vector_type(8)))  float         v8f;
typedef __attribute__((ext_vector_type(4))) unsigned int   v4u;
typedef __attribute__((ext_vector_type(4)))  float         v4f;

// global (address_space 1) views for weight loads -> global_load_b128, LOADcnt only
typedef __attribute__((address_space(1))) const unsigned short* gushortp;
typedef __attribute__((address_space(1))) const v4u*            gv4up;

#define SEQ   512
#define HID   256
#define BATCH 512
#define OUTN  10
#define BN    32      // batch tile per workgroup
#define KAUG  288     // augmented K: 256 (Wh) + 1 (Wx) + 1 (bias) + 30 zero pad
#define NK8   9       // K tiles of 32
#define HSTR  296     // padded LDS stride (592B row -> 20-bank lane stride, conflict-free)

static __device__ __forceinline__ unsigned short f2bf(float f) {
    unsigned int u = __float_as_uint(f);
    u += 0x7FFFu + ((u >> 16) & 1u);           // round-to-nearest-even
    return (unsigned short)(u >> 16);
}
static __device__ __forceinline__ float sigf(float x) {
    return __builtin_amdgcn_rcpf(1.f + __expf(-x));
}
static __device__ __forceinline__ float tanhf_(float x) {
    return 2.f * sigf(2.f * x) - 1.f;
}

// ---- 1) build augmented bf16 weights: [ W_*h | W_*x | b_* | 0...0 ], row stride 288 ----
__global__ void prep_weights(const float* __restrict__ Wg, const float* __restrict__ Wi,
                             const float* __restrict__ Wf, const float* __restrict__ Wo,
                             const float* __restrict__ Xg, const float* __restrict__ Xi,
                             const float* __restrict__ Xf, const float* __restrict__ Xo,
                             const float* __restrict__ Bg, const float* __restrict__ Bi,
                             const float* __restrict__ Bf, const float* __restrict__ Bo,
                             unsigned short* __restrict__ waug) {
    int k = blockIdx.x * 32 + threadIdx.x;   // 0..287
    int m = blockIdx.y;                      // 0..255
    int g = blockIdx.z;                      // 0..3
    const float* Wh = (g == 0) ? Wg : (g == 1) ? Wi : (g == 2) ? Wf : Wo;
    const float* Wx = (g == 0) ? Xg : (g == 1) ? Xi : (g == 2) ? Xf : Xo;
    const float* Bb = (g == 0) ? Bg : (g == 1) ? Bi : (g == 2) ? Bf : Bo;
    float v = 0.f;
    if (k < HID)           v = Wh[m * HID + k];
    else if (k == HID)     v = Wx[m];
    else if (k == HID + 1) v = Bb[m];
    waug[((size_t)g * HID + m) * KAUG + k] = f2bf(v);
}

// ---- 2) persistent per-batch-tile LSTM scan, WMMA bf16, augmented GEMM ----
__global__ void __launch_bounds__(256)
__attribute__((amdgpu_waves_per_eu(2)))
lstm_kernel(const float* __restrict__ x,
            const unsigned short* __restrict__ waug,
            float* __restrict__ hout) {
    __shared__ __align__(16) unsigned short hT[2][BN][HSTR]; // bf16 [h | x | 1 | 0pad], batch-major, dbl-buffered

    const int tid       = threadIdx.x;
    const int lane      = tid & 31;
    const int wave      = tid >> 5;
    const int batchBase = blockIdx.x * BN;

    unsigned short* hb0 = &hT[0][0][0];
    unsigned short* hb1 = &hT[1][0][0];

    for (int i = tid; i < 2 * BN * HSTR; i += 256) ((unsigned short*)hT)[i] = 0;
    if (tid < 32) {   // seed augmented rows for t=0: row 256 = x[:,0], row 257 = 1.0
        float xn = x[(size_t)(batchBase + tid) * SEQ];
        *(unsigned int*)(hb0 + tid * HSTR + HID) =
            (unsigned int)f2bf(xn) | (0x3F80u << 16);
    }
    __syncthreads();

    const int nlo = lane & 15;          // column within 16-wide n-tile
    const int hi  = (lane >> 4) & 1;    // lane half

    // per-lane invariant base pointers; loop indices become immediate offsets
    const int rOff = nlo * HSTR + hi * 8;              // B read:  + nt*16*HSTR + kT
    const int wOff = nlo * HSTR + hi * 8 + wave * 32;  // h write: + nt*16*HSTR + mi*16

    float c[2][2][8];                   // persistent cell state (fixed tile ownership)
    #pragma unroll
    for (int a = 0; a < 2; ++a)
        #pragma unroll
        for (int b = 0; b < 2; ++b)
            #pragma unroll
            for (int r = 0; r < 8; ++r) c[a][b][r] = 0.f;

    // weight base pointer kept in address space 1 -> global_load_b128 (LOADcnt only);
    // opaque per-iteration redefinition blocks LICM (no hoist-then-spill).
    gushortp awt = (gushortp)(waug + ((size_t)(wave * 32 + nlo) * KAUG + hi * 8));

    for (int t = 0; t < SEQ; ++t) {
        asm volatile("" : "+v"(awt));   // keep weight loads inside the t-loop
        const int cur = t & 1;
        const unsigned short* rB = (cur ? hb1 : hb0) + rOff;
        unsigned short*       wB = (cur ? hb0 : hb1) + wOff;

        #pragma unroll
        for (int mi = 0; mi < 2; ++mi) {
            v8f acc[4][2];
            #pragma unroll
            for (int g = 0; g < 4; ++g) { acc[g][0] = {}; acc[g][1] = {}; }

            #pragma unroll
            for (int k8 = 0; k8 < NK8; ++k8) {
                const int kT = k8 * 32;

                union { v16bf v; v4u u[2]; } bfr[2];
                #pragma unroll
                for (int nt = 0; nt < 2; ++nt) {
                    const unsigned short* bp = rB + nt * 16 * HSTR + kT;
                    bfr[nt].u[0] = *(const v4u*)(bp);
                    bfr[nt].u[1] = *(const v4u*)(bp + 16);
                }
                #pragma unroll
                for (int g = 0; g < 4; ++g) {
                    union { v16bf v; v4u u[2]; } afr;
                    gushortp ap = awt + ((size_t)g * HID * KAUG + mi * 16 * KAUG + kT);
                    afr.u[0] = *(gv4up)(ap);
                    afr.u[1] = *(gv4up)(ap + 16);
                    acc[g][0] = __builtin_amdgcn_wmma_f32_16x16x32_bf16(
                        false, afr.v, false, bfr[0].v, (short)0, acc[g][0], false, false);
                    acc[g][1] = __builtin_amdgcn_wmma_f32_16x16x32_bf16(
                        false, afr.v, false, bfr[1].v, (short)0, acc[g][1], false, false);
                }
            }

            // activations + state update in D layout (bias & x already inside acc)
            #pragma unroll
            for (int nt = 0; nt < 2; ++nt) {
                float hv[8];
                #pragma unroll
                for (int r = 0; r < 8; ++r) {
                    float gg = tanhf_(acc[0][nt][r]);
                    float ii = sigf(acc[1][nt][r]);
                    float ff = sigf(acc[2][nt][r]);
                    float oo = sigf(acc[3][nt][r]);
                    float cn = gg * ii + c[mi][nt][r] * ff;
                    c[mi][nt][r] = cn;
                    hv[r] = tanhf_(cn) * oo;
                }
                v4u pk;
                #pragma unroll
                for (int p = 0; p < 4; ++p)
                    pk[p] = (unsigned int)f2bf(hv[2 * p]) |
                            ((unsigned int)f2bf(hv[2 * p + 1]) << 16);
                *(v4u*)(wB + nt * 16 * HSTR + mi * 16) = pk;

                if (t == SEQ - 1) {
                    float* hp = hout + (size_t)(batchBase + nt * 16 + nlo) * HID
                              + (wave * 32 + mi * 16 + hi * 8);
                    v4f lo  = { hv[0], hv[1], hv[2], hv[3] };
                    v4f hi4 = { hv[4], hv[5], hv[6], hv[7] };
                    *(v4f*)hp       = lo;
                    *(v4f*)(hp + 4) = hi4;
                }
            }
        }

        if (tid < 32) {   // carry x forward: row 256 = x[:,t+1], row 257 = 1.0
            int tp = (t + 1) & (SEQ - 1);             // last-step write is unused, stays in-bounds
            float xn = x[(size_t)(batchBase + tid) * SEQ + tp];
            *(unsigned int*)((cur ? hb0 : hb1) + tid * HSTR + HID) =
                (unsigned int)f2bf(xn) | (0x3F80u << 16);
        }
        __syncthreads();
    }
}

// ---- 3) projection (10x256) + softmax, one thread per batch column ----
__global__ void proj_softmax(const float* __restrict__ hfin,
                             const float* __restrict__ Wp, const float* __restrict__ bp,
                             float* __restrict__ out) {
    int n = blockIdx.x * blockDim.x + threadIdx.x;
    if (n >= BATCH) return;
    const float* h = hfin + (size_t)n * HID;
    float logit[OUTN];
    float mx = -1e30f;
    #pragma unroll
    for (int o = 0; o < OUTN; ++o) {
        const float* w = Wp + o * HID;
        float s = bp[o];
        for (int k = 0; k < HID; k += 4) {
            v4f wv = *(const v4f*)(w + k);
            v4f hv = *(const v4f*)(h + k);
            s += wv[0] * hv[0] + wv[1] * hv[1] + wv[2] * hv[2] + wv[3] * hv[3];
        }
        logit[o] = s;
        mx = fmaxf(mx, s);
    }
    float den = 0.f;
    #pragma unroll
    for (int o = 0; o < OUTN; ++o) { float e = __expf(logit[o] - mx); logit[o] = e; den += e; }
    float inv = 1.f / den;
    #pragma unroll
    for (int o = 0; o < OUTN; ++o) out[(size_t)n * OUTN + o] = logit[o] * inv;
}

extern "C" void kernel_launch(void* const* d_in, const int* in_sizes, int n_in,
                              void* d_out, int out_size, void* d_ws, size_t ws_size,
                              hipStream_t stream) {
    const float* x   = (const float*)d_in[0];
    const float* Wgx = (const float*)d_in[1];
    const float* Wgh = (const float*)d_in[2];
    const float* bg  = (const float*)d_in[3];
    const float* Wix = (const float*)d_in[4];
    const float* Wih = (const float*)d_in[5];
    const float* bi  = (const float*)d_in[6];
    const float* Wfx = (const float*)d_in[7];
    const float* Wfh = (const float*)d_in[8];
    const float* bf  = (const float*)d_in[9];
    const float* Wox = (const float*)d_in[10];
    const float* Woh = (const float*)d_in[11];
    const float* bo  = (const float*)d_in[12];
    const float* Wph = (const float*)d_in[13];
    const float* bp  = (const float*)d_in[14];

    unsigned short* waug = (unsigned short*)d_ws;                       // 4*256*288 bf16 = 576 KB
    float* hfin = (float*)((char*)d_ws + (size_t)4 * HID * KAUG * 2);   // 512*256 f32   = 512 KB

    prep_weights<<<dim3(KAUG / 32, HID, 4), dim3(32), 0, stream>>>(
        Wgh, Wih, Wfh, Woh, Wgx, Wix, Wfx, Wox, bg, bi, bf, bo, waug);
    lstm_kernel<<<dim3(BATCH / BN), dim3(256), 0, stream>>>(x, waug, hfin);
    proj_softmax<<<dim3(BATCH / 256), dim3(256), 0, stream>>>(hfin, Wph, bp, (float*)d_out);
}
